// GraphLinearDeform_72060961292436
// MI455X (gfx1250) — matrix operand order repaired and checked by reading, not compiled, not run on hardware
//
#include <hip/hip_runtime.h>
#include <hip/hip_bf16.h>
#include <math.h>

typedef __attribute__((ext_vector_type(16))) _Float16 v16h;
typedef __attribute__((ext_vector_type(8)))  _Float16 v8h;
typedef __attribute__((ext_vector_type(8)))  float    v8f;

#define EPSV 1e-5f
#define NEG_SLOPE 0.3f
#define MPTS 120000
#define STRIDE 136       // halfs per LDS activation row (128 + 8 pad -> 272B, bank-rotating)
#define PTS 64           // points per block (4 WMMA N-tiles)
#define NT  (PTS / 16)

// ---- workspace layout (bytes) ----
// frags: 56 fragments * 32 lanes * 16 halfs * 2B = 57344
//   layer1: frag[0..7]    (ot)        K=32 (padded from 8)
//   layer2: frag[8..39]   (ot*4+kt)   K=128
//   layer3: frag[40..55]  (ot*4+kt)   K=128, out=64
#define FRAG_BYTES   57344
#define BIAS1_OFF    57344          // 128 f32
#define BIAS2_OFF    (BIAS1_OFF + 128*4)
#define BIAS3_OFF    (BIAS2_OFF + 128*4)   // 64 f32
#define VBUF0_OFF    58880
#define VBUF1_OFF    (VBUF0_OFF + 720000*4)

// ---------------------------------------------------------------------------
// Prep: fold BatchNorm into weights/bias and pre-swizzle weights into the
// exact per-lane WMMA A-fragment order (ISA 16-bit A 16x32 layout).
// ---------------------------------------------------------------------------
__global__ void __launch_bounds__(256) prep_kernel(
    const float* __restrict__ W1, const float* __restrict__ b1, const float* __restrict__ beta1,
    const float* __restrict__ gamma1, const float* __restrict__ mean1, const float* __restrict__ var1,
    const float* __restrict__ W2, const float* __restrict__ b2, const float* __restrict__ beta2,
    const float* __restrict__ gamma2, const float* __restrict__ mean2, const float* __restrict__ var2,
    const float* __restrict__ W3, const float* __restrict__ b3, const float* __restrict__ beta3,
    const float* __restrict__ gamma3, const float* __restrict__ mean3, const float* __restrict__ var3,
    _Float16* __restrict__ frags, float* __restrict__ bias1, float* __restrict__ bias2,
    float* __restrict__ bias3)
{
    int gtid = blockIdx.x * 256 + threadIdx.x;

    // folded biases: t_o = s_o*(b_o - mean_o) + beta_o, s_o = gamma_o*rsqrt(var_o+eps)
    if (gtid < 128) {
        float s = gamma1[gtid] * rsqrtf(var1[gtid] + EPSV);
        bias1[gtid] = s * (b1[gtid] - mean1[gtid]) + beta1[gtid];
    } else if (gtid < 256) {
        int o = gtid - 128;
        float s = gamma2[o] * rsqrtf(var2[o] + EPSV);
        bias2[o] = s * (b2[o] - mean2[o]) + beta2[o];
    } else if (gtid < 320) {
        int o = gtid - 256;
        float s = gamma3[o] * rsqrtf(var3[o] + EPSV);
        bias3[o] = s * (b3[o] - mean3[o]) + beta3[o];
    }

    if (gtid >= 56 * 32) return;          // one thread per (fragment, lane)
    int f    = gtid >> 5;
    int lane = gtid & 31;
    int M    = lane & 15;
    int half = lane >> 4;

    const float *W, *gm, *vr;
    int Kreal, o, kt;
    if (f < 8)       { W = W1; gm = gamma1; vr = var1; Kreal = 8;   o = 16 * f + M;               kt = 0;            }
    else if (f < 40) { int ff = f - 8;  W = W2; gm = gamma2; vr = var2; Kreal = 128; o = 16 * (ff >> 2) + M; kt = ff & 3; }
    else             { int ff = f - 40; W = W3; gm = gamma3; vr = var3; Kreal = 128; o = 16 * (ff >> 2) + M; kt = ff & 3; }

    float s = gm[o] * rsqrtf(vr[o] + EPSV);
    _Float16* dst = frags + (size_t)gtid * 16;
#pragma unroll
    for (int j = 0; j < 16; ++j) {
        // 16-bit A-matrix 16x32 lane layout: K = 16*(j>>3) + 8*half + (j&7)
        int K = 32 * kt + ((j >> 3) << 4) + (half << 3) + (j & 7);
        float v = (K < Kreal) ? s * W[o * Kreal + K] : 0.f;
        dst[j] = (_Float16)v;
    }
}

// ---------------------------------------------------------------------------
// Fused per-iteration kernel: grid_sample3d -> 3x WMMA MLP layers -> final
// 3x64 projection -> vertex update.  Block = 256 thr (8 waves) = 64 points.
// Wave wv owns output-channel tile [16wv,16wv+16) and iterates 4 N-tiles,
// reusing its A-fragments (weights) held in VGPRs across all 4.
// ---------------------------------------------------------------------------
__device__ __forceinline__ v8f wmma_f16(v16h a, v16h b, v8f c) {
    return __builtin_amdgcn_wmma_f32_16x16x32_f16(false, a, false, b, (short)0, c, false, false);
}

__device__ __forceinline__ v16h load_bfrag(const _Float16* buf, int pt, int half, int kt) {
    // B 32x16 layout: lane (N = lane&15); lanes 0-15 hold K=0..7,16..23, lanes 16-31 K=8..15,24..31
    const _Float16* p = buf + pt * STRIDE + kt * 32 + half * 8;
    v8h lo = *(const v8h*)(p);
    v8h hi = *(const v8h*)(p + 16);
    return __builtin_shufflevector(lo, hi, 0,1,2,3,4,5,6,7,8,9,10,11,12,13,14,15);
}

__device__ __forceinline__ void store_act(_Float16* buf, v8f c, int pt, int half, int wv,
                                          const float* __restrict__ bias) {
    int chb = wv * 16 + half * 8;       // D layout: M = v + 8*half, ch = 16*wv + M
    v8h h;
#pragma unroll
    for (int v = 0; v < 8; ++v) {
        float f = c[v] + bias[chb + v];
        f = (f > 0.f) ? f : NEG_SLOPE * f;   // LeakyReLU
        h[v] = (_Float16)f;
    }
    *(v8h*)(buf + pt * STRIDE + chb) = h;    // one ds_store_b128
}

__global__ void __launch_bounds__(256) deform_iter_kernel(
    const float* __restrict__ image, const float* __restrict__ vin, float* __restrict__ vout,
    const _Float16* __restrict__ frags,
    const float* __restrict__ bias1, const float* __restrict__ bias2, const float* __restrict__ bias3,
    const float* __restrict__ Wf, const float* __restrict__ bf)
{
    __shared__ __align__(32) _Float16 buf0[PTS * STRIDE];
    __shared__ __align__(32) _Float16 buf1[PTS * STRIDE];

    const int tid    = threadIdx.x;
    const int block0 = blockIdx.x * PTS;     // first global point of this tile

    // zero K-padding (channels 8..31) for the K=32 first layer
    for (int i = tid; i < PTS * 24; i += 256) {
        int pt = i / 24, c = 8 + i % 24;
        buf0[pt * STRIDE + c] = (_Float16)0.f;
    }

    // ---- trilinear grid sample: task = (point, channel), 2 tasks/thread ----
    for (int i = tid; i < PTS * 8; i += 256) {
        int pt = i >> 3, ch = i & 7;
        int gp = block0 + pt;
        int n  = gp / MPTS;
        float gx = vin[gp * 3 + 0], gy = vin[gp * 3 + 1], gz = vin[gp * 3 + 2];
        float x = (gx + 1.f) * 48.f - 0.5f;
        float y = (gy + 1.f) * 48.f - 0.5f;
        float z = (gz + 1.f) * 48.f - 0.5f;
        float x0f = floorf(x), y0f = floorf(y), z0f = floorf(z);
        int   x0 = (int)x0f, y0 = (int)y0f, z0 = (int)z0f;
        float wx1 = x - x0f, wy1 = y - y0f, wz1 = z - z0f;
        float wx0 = 1.f - wx1, wy0 = 1.f - wy1, wz0 = 1.f - wz1;
        const float* img = image + (size_t)(n * 8 + ch) * 884736;   // 96^3
        float acc = 0.f;
#pragma unroll
        for (int cor = 0; cor < 8; ++cor) {
            int dx = cor & 1, dy = (cor >> 1) & 1, dz = cor >> 2;
            int xi = x0 + dx, yi = y0 + dy, zi = z0 + dz;
            float w = (dx ? wx1 : wx0) * (dy ? wy1 : wy0) * (dz ? wz1 : wz0);
            bool ok = (xi >= 0) & (xi < 96) & (yi >= 0) & (yi < 96) & (zi >= 0) & (zi < 96);
            int xc = min(max(xi, 0), 95), yc = min(max(yi, 0), 95), zc = min(max(zi, 0), 95);
            float v = img[(zc * 96 + yc) * 96 + xc];
            acc += ok ? w * v : 0.f;
        }
        buf0[pt * STRIDE + ch] = (_Float16)acc;
    }
    __syncthreads();

    const int wv = tid >> 5, lane = tid & 31;
    const int ptl = lane & 15, half = lane >> 4;

    // ---- layer 1: 8(->32 pad) -> 128; A loaded once, reused over 4 N-tiles ----
    {
        v16h a = *(const v16h*)(frags + (size_t)(wv * 32 + lane) * 16);
#pragma unroll
        for (int nt = 0; nt < NT; ++nt) {
            int pt = nt * 16 + ptl;
            v16h b = load_bfrag(buf0, pt, half, 0);
            v8f  c = {};
            c = wmma_f16(a, b, c);
            store_act(buf1, c, pt, half, wv, bias1);
        }
    }
    __syncthreads();

    // ---- layer 2: 128 -> 128 ----
    {
        v16h a[4];
#pragma unroll
        for (int kt = 0; kt < 4; ++kt)
            a[kt] = *(const v16h*)(frags + (size_t)((8 + wv * 4 + kt) * 32 + lane) * 16);
#pragma unroll
        for (int nt = 0; nt < NT; ++nt) {
            int pt = nt * 16 + ptl;
            v8f c = {};
#pragma unroll
            for (int kt = 0; kt < 4; ++kt)
                c = wmma_f16(a[kt], load_bfrag(buf1, pt, half, kt), c);
            store_act(buf0, c, pt, half, wv, bias2);
        }
    }
    __syncthreads();

    // ---- layer 3: 128 -> 64 (waves 0..3) ----
    if (wv < 4) {
        v16h a[4];
#pragma unroll
        for (int kt = 0; kt < 4; ++kt)
            a[kt] = *(const v16h*)(frags + (size_t)((40 + wv * 4 + kt) * 32 + lane) * 16);
#pragma unroll
        for (int nt = 0; nt < NT; ++nt) {
            int pt = nt * 16 + ptl;
            v8f c = {};
#pragma unroll
            for (int kt = 0; kt < 4; ++kt)
                c = wmma_f16(a[kt], load_bfrag(buf0, pt, half, kt), c);
            store_act(buf1, c, pt, half, wv, bias3);
        }
    }
    __syncthreads();

    // ---- final 3x64 projection + vertex update: task = (point, coord) ----
    for (int t = tid; t < PTS * 3; t += 256) {
        int p = t / 3, o = t % 3;
        const _Float16* h = buf1 + p * STRIDE;
        float acc = bf[o];
#pragma unroll 8
        for (int c2 = 0; c2 < 64; ++c2) acc += Wf[o * 64 + c2] * (float)h[c2];
        int gp = block0 + p;
        vout[gp * 3 + o] = vin[gp * 3 + o] + (0.01f / 3.0f) * acc;
    }
}

// ---------------------------------------------------------------------------
extern "C" void kernel_launch(void* const* d_in, const int* in_sizes, int n_in,
                              void* d_out, int out_size, void* d_ws, size_t ws_size,
                              hipStream_t stream) {
    // locate inputs by unique flat size (robust to flatten-order details);
    // within a layer the pytree order is W, b, beta, gamma, mean, var.
    auto find_sz = [&](int sz) -> int {
        for (int i = 0; i < n_in; ++i) if (in_sizes[i] == sz) return i;
        return 0;
    };
    int iImg = find_sz(2 * 8 * 96 * 96 * 96);
    int iV   = find_sz(720000);
    int iWf  = find_sz(3 * 64);
    int iBf  = find_sz(3);
    int iW1  = find_sz(128 * 8);
    int iW2  = find_sz(128 * 128);
    int iW3  = find_sz(64 * 128);

    const float* image = (const float*)d_in[iImg];
    const float* verts = (const float*)d_in[iV];
    const float* Wf = (const float*)d_in[iWf];
    const float* bf = (const float*)d_in[iBf];
    const float *W1 = (const float*)d_in[iW1],      *b1 = (const float*)d_in[iW1 + 1],
                *be1 = (const float*)d_in[iW1 + 2], *g1 = (const float*)d_in[iW1 + 3],
                *m1 = (const float*)d_in[iW1 + 4],  *v1 = (const float*)d_in[iW1 + 5];
    const float *W2 = (const float*)d_in[iW2],      *b2 = (const float*)d_in[iW2 + 1],
                *be2 = (const float*)d_in[iW2 + 2], *g2 = (const float*)d_in[iW2 + 3],
                *m2 = (const float*)d_in[iW2 + 4],  *v2 = (const float*)d_in[iW2 + 5];
    const float *W3 = (const float*)d_in[iW3],      *b3 = (const float*)d_in[iW3 + 1],
                *be3 = (const float*)d_in[iW3 + 2], *g3 = (const float*)d_in[iW3 + 3],
                *m3 = (const float*)d_in[iW3 + 4],  *v3 = (const float*)d_in[iW3 + 5];

    char* ws = (char*)d_ws;
    _Float16* frags = (_Float16*)(ws);
    float* bias1 = (float*)(ws + BIAS1_OFF);
    float* bias2 = (float*)(ws + BIAS2_OFF);
    float* bias3 = (float*)(ws + BIAS3_OFF);
    float* vbuf0 = (float*)(ws + VBUF0_OFF);
    float* vbuf1 = (float*)(ws + VBUF1_OFF);
    float* vout  = (float*)d_out;

    // one-time per launch: fold BN + swizzle weights into WMMA A-fragments
    prep_kernel<<<7, 256, 0, stream>>>(W1, b1, be1, g1, m1, v1,
                                       W2, b2, be2, g2, m2, v2,
                                       W3, b3, be3, g3, m3, v3,
                                       frags, bias1, bias2, bias3);

    const int nblocks = 240000 / PTS;   // 64 points per block
    deform_iter_kernel<<<nblocks, 256, 0, stream>>>(image, verts, vbuf0, frags,
                                                    bias1, bias2, bias3, Wf, bf);
    deform_iter_kernel<<<nblocks, 256, 0, stream>>>(image, vbuf0, vbuf1, frags,
                                                    bias1, bias2, bias3, Wf, bf);
    deform_iter_kernel<<<nblocks, 256, 0, stream>>>(image, vbuf1, vout, frags,
                                                    bias1, bias2, bias3, Wf, bf);
}
